// MultiScaleDeformConv_3x3_64914135712266
// MI455X (gfx1250) — matrix operand order repaired and checked by reading, not compiled, not run on hardware
//
#include <hip/hip_runtime.h>
#include <hip/hip_bf16.h>
#include <math.h>

// ---------------------------------------------------------------------------
// MultiScaleDeformConv for MI455X (gfx1250, wave32, WMMA).
// All dense convolutions (offset nets K=3/5/7 and the 1x1 pointwise convs)
// run as implicit-im2col GEMMs on v_wmma_f32_16x16x32_bf16 (f32 accumulate).
//  * im2col K dimension is TAP-MAJOR (k = t*128 + c): one tap / 32-wide K step
//    -> uniform (dy,dx) + single bounds check per step, no divides in the loop.
//  * Activations are kept channels-last (NHWC bf16): a lane's 16-channel A
//    fragment is two contiguous 16B chunks -> 2x global_load_b128 per subtile.
//  * B matrices pre-swizzled to WMMA fragment order: LDS staging is contiguous
//    uint4 copies; each lane reads its fragment as 2x ds_load_b128.
//  * Each wave computes MW=2 M-subtiles to double WMMA work per B fragment.
// Per-group working set (~50MB) fits the 192MB L2. Workspace ~125 MB.
// ---------------------------------------------------------------------------

constexpr int NB  = 8;     // batch
constexpr int CT  = 384;   // total channels
constexpr int CS  = 128;   // channels per group
constexpr int HH  = 96;
constexpr int WW  = 96;
constexpr int HWL = HH * WW;
constexpr int NTOT = NB * CT * HWL;          // 28,311,552
constexpr int MROWS = NB * HWL;              // 73,728  (GEMM M)

typedef __attribute__((ext_vector_type(16))) __bf16          v16bf;
typedef __attribute__((ext_vector_type(16))) unsigned short  v16u;
typedef __attribute__((ext_vector_type(8)))  float           v8f;

struct U32x8 { uint4 a, b; };   // 32B carrier for a 16-half fragment

__device__ __forceinline__ unsigned short f2bf(float f) {
    unsigned int u = __float_as_uint(f);
    u += 0x7FFFu + ((u >> 16) & 1u);          // round-to-nearest-even
    return (unsigned short)(u >> 16);
}
__device__ __forceinline__ unsigned short to_bf(float f)          { return f2bf(f); }
__device__ __forceinline__ unsigned short to_bf(unsigned short u) { return u; }

// ----------------------- NCHW -> NHWC (bf16) transpose ---------------------
// src: [NB][CC][HWL] of T (float or bf16-bits); dst: [NB][HWL][CC] bf16 bits.
// Tile = 32 channels x 64 pixels. Loads coalesced along pixels; stores are
// uint4 (8 channels) per thread, coalesced along channels.

template <typename T, int CC>
__global__ __launch_bounds__(256)
void nchw_to_nhwc_bf16(const T* __restrict__ src, unsigned short* __restrict__ dst) {
    constexpr int TPP = HWL / 64;            // pixel tiles per image plane
    constexpr int TPC = CC / 32;             // channel tiles
    __shared__ unsigned short sT[32][65];    // +1 pad against bank conflicts

    int tile = blockIdx.x;
    int b  = tile / (TPC * TPP);
    int r  = tile - b * (TPC * TPP);
    int ct = r / TPP;
    int pt = r - ct * TPP;

    const T* s = src + ((size_t)b * CC + ct * 32) * HWL + pt * 64;
    for (int e = threadIdx.x; e < 32 * 64; e += 256) {
        int c = e >> 6, p = e & 63;
        sT[c][p] = to_bf(s[(size_t)c * HWL + p]);
    }
    __syncthreads();

    int p  = threadIdx.x >> 2;
    int c8 = (threadIdx.x & 3) * 8;
    unsigned short tmp[8];
#pragma unroll
    for (int q = 0; q < 8; ++q) tmp[q] = sT[c8 + q][p];
    unsigned short* d = dst + ((size_t)b * HWL + pt * 64 + p) * CC + ct * 32 + c8;
    *(uint4*)d = *(const uint4*)tmp;
}

// Repack conv weights w[n][c*K2+t] into WMMA-fragment-ordered bf16:
//   Bs[((s*NT + nt)*32 + l)*16 + h], k = s*32 + (l>>4)*8 + (h&7) + (h>>3)*16 (tap-major
//   k = t*128 + c), n = nt*16 + (l&15), zero padded for n >= Nreal.
__global__ __launch_bounds__(256) void cvt_wB_swz(const float* __restrict__ w,
                                                  unsigned short* __restrict__ Bs,
                                                  int Kdim, int K2, int NT, int Nreal) {
    int i = blockIdx.x * 256 + threadIdx.x;
    if (i >= Kdim * NT * 16) return;
    int h  = i & 15;
    int l  = (i >> 4) & 31;
    int nt = (i >> 9) % NT;
    int s  = i / (512 * NT);
    int k  = s * 32 + (l >> 4) * 8 + (h & 7) + (h >> 3) * 16;     // tap-major k
    int n  = nt * 16 + (l & 15);
    int c  = k & 127;                        // CS = 128
    int t  = k >> 7;
    Bs[i] = (n < Nreal) ? f2bf(w[n * Kdim + c * K2 + t]) : (unsigned short)0;
}

// -------------------- implicit-im2col GEMM on WMMA -------------------------
// xin: NHWC bf16, Cchan channels per pixel, group channel base cOff.
// out[b, outCbase+n, y, x] (NCHW f32) = bias[n] + sum_{t,c} A*W.

template <int KK, int NT>
__global__ __launch_bounds__(256)
void conv_gemm_wmma(const unsigned short* __restrict__ xin, int Cchan, int cOff,
                    const unsigned short* __restrict__ Bs,
                    const float* __restrict__ bias,
                    float* __restrict__ out, int outCtot, int outCbase, int Nreal) {
    constexpr int K2   = KK * KK;
    constexpr int KDIM = CS * K2;           // divisible by 32
    constexpr int NPAD = NT * 16;
    constexpr int PAD  = KK / 2;
    constexpr int MW   = 2;                 // M subtiles per wave
    __shared__ unsigned short sB[NT * 32 * 16];   // fragment-ordered K-step tile

    const int tid   = threadIdx.x;
    const int lane  = tid & 31;
    const int wave  = tid >> 5;
    const int mBase = blockIdx.x * (128 * MW) + wave * (16 * MW);

    const int hiSel = lane >> 4;
    const int aKOff = hiSel * 8;            // A-layout channel offset
    const int nLane = lane & 15;

    int yIdx[MW], xIdx[MW];
    long rowBase[MW];                        // NHWC pixel base (element index)
#pragma unroll
    for (int mw = 0; mw < MW; ++mw) {
        int mRow = mBase + mw * 16 + nLane;
        int bI   = mRow / HWL;
        int rem  = mRow - bI * HWL;
        yIdx[mw] = rem / WW;
        xIdx[mw] = rem - yIdx[mw] * WW;
        rowBase[mw] = (long)mRow * Cchan;    // (b*HWL + y*W + x) * Cchan
    }

    v8f acc[MW][NT];
#pragma unroll
    for (int mw = 0; mw < MW; ++mw)
        for (int t = 0; t < NT; ++t)
            for (int r = 0; r < 8; ++r) acc[mw][t][r] = 0.0f;

    for (int kstep = 0; kstep < KDIM; kstep += 32) {
        // Stage fragment-ordered B K-step tile: contiguous, uint4-wide.
        {
            const uint4* src = (const uint4*)(Bs + kstep * NPAD);
            uint4* dst = (uint4*)sB;
            for (int e = tid; e < NT * 64; e += 256) dst[e] = src[e];
        }
        __syncthreads();

        const int t  = kstep >> 7;          // single tap for this K step
        const int dy = t / KK - PAD;
        const int dx = t % KK - PAD;
        // uniform scalar pointer shift for this K step
        const long cShift = (long)(dy * WW + dx) * Cchan + (kstep & 127) + cOff + aKOff;

        v16bf afrag[MW];
#pragma unroll
        for (int mw = 0; mw < MW; ++mw) {
            int yy = yIdx[mw] + dy;
            int xx = xIdx[mw] + dx;
            v16u au;
            if (yy >= 0 && yy < HH && xx >= 0 && xx < WW) {
                const unsigned short* p = xin + rowBase[mw] + cShift;
                U32x8 raw;
                raw.a = *(const uint4*)p;            // channels +0..7
                raw.b = *(const uint4*)(p + 16);     // channels +16..23
                au = __builtin_bit_cast(v16u, raw);
            } else {
#pragma unroll
                for (int q = 0; q < 16; ++q) au[q] = 0;
            }
            afrag[mw] = __builtin_bit_cast(v16bf, au);
        }

#pragma unroll
        for (int nt = 0; nt < NT; ++nt) {
            // Whole fragment is 32 contiguous bytes in LDS -> 2x ds_load_b128.
            const v16u* bp = (const v16u*)(sB + (nt * 32 + lane) * 16);
            v16bf bfrag = __builtin_bit_cast(v16bf, *bp);
#pragma unroll
            for (int mw = 0; mw < MW; ++mw)
                acc[mw][nt] = __builtin_amdgcn_wmma_f32_16x16x32_bf16(
                    false, afrag[mw], false, bfrag, (short)0, acc[mw][nt], false, false);
        }
        __syncthreads();
    }

    // C/D layout: lanes 0-15 -> N=lane, M=r; lanes 16-31 -> N=lane-16, M=r+8.
#pragma unroll
    for (int nt = 0; nt < NT; ++nt) {
        int n = nt * 16 + nLane;
        if (n >= Nreal) continue;
        float bv = bias[n];
#pragma unroll
        for (int mw = 0; mw < MW; ++mw) {
#pragma unroll
            for (int r = 0; r < 8; ++r) {
                int m  = mBase + mw * 16 + r + hiSel * 8;
                int b2 = m / HWL;
                int r2 = m - b2 * HWL;
                int y2 = r2 / WW;
                int x2 = r2 - y2 * WW;
                out[((b2 * outCtot + outCbase + n) * HH + y2) * WW + x2] =
                    acc[mw][nt][r] + bv;
            }
        }
    }
}

// ------------------- deformable depthwise (bilinear) -----------------------
// Reads NCHW f32 x (coalesced along x), writes NCHW bf16; a cheap tiled
// transpose flips the result to NHWC for the pointwise GEMM.

template <int KK>
__global__ __launch_bounds__(256)
void deform_dw(const float* __restrict__ x, int xCtot, int xCbase,
               const float* __restrict__ offs,   // [NB, 2*K*K, H, W]
               const float* __restrict__ dw,     // [CS, K, K]
               unsigned short* __restrict__ outb) {
    constexpr int PAD = KK / 2;
    constexpr int K2  = KK * KK;
    int idx = blockIdx.x * 256 + threadIdx.x;
    if (idx >= NB * CS * HWL) return;
    int xw = idx % WW;
    int r  = idx / WW;
    int y  = r % HH;  r /= HH;
    int c  = r % CS;
    int b  = r / CS;

    const float* xp = x + (size_t)(b * xCtot + xCbase + c) * HWL;
    const float* op = offs + (size_t)b * (2 * K2) * HWL + (size_t)y * WW + xw;

    float acc = 0.0f;
    for (int t = 0; t < K2; ++t) {
        float ody = op[(size_t)(2 * t) * HWL];
        float odx = op[(size_t)(2 * t + 1) * HWL];
        float py = (float)(y + t / KK - PAD) + ody;
        float px = (float)(xw + t % KK - PAD) + odx;
        float y0 = floorf(py), x0 = floorf(px);
        float fy = py - y0,    fx = px - x0;
        int iy = (int)y0, ix = (int)x0;
        float v = 0.0f;
        if (iy >= 0     && iy < HH     && ix >= 0     && ix < WW)
            v += (1.f - fy) * (1.f - fx) * xp[iy * WW + ix];
        if (iy >= 0     && iy < HH     && ix + 1 >= 0 && ix + 1 < WW)
            v += (1.f - fy) * fx * xp[iy * WW + ix + 1];
        if (iy + 1 >= 0 && iy + 1 < HH && ix >= 0     && ix < WW)
            v += fy * (1.f - fx) * xp[(iy + 1) * WW + ix];
        if (iy + 1 >= 0 && iy + 1 < HH && ix + 1 >= 0 && ix + 1 < WW)
            v += fy * fx * xp[(iy + 1) * WW + ix + 1];
        acc += v * dw[c * K2 + t];
    }
    outb[idx] = f2bf(acc);
}

// ------------------------------- SE gate -----------------------------------

__global__ __launch_bounds__(256) void se_pool(const float* __restrict__ out,
                                               float* __restrict__ pooled) {
    // pooled[0..3071] = avg, pooled[3072..6143] = max
    int bc = blockIdx.x;
    const float* p = out + (size_t)bc * HWL;
    float s = 0.0f, mx = -3.402823e38f;
    for (int i = threadIdx.x; i < HWL; i += 256) {
        float v = p[i];
        s += v;
        mx = fmaxf(mx, v);
    }
    __shared__ float ss[256], sm[256];
    ss[threadIdx.x] = s; sm[threadIdx.x] = mx;
    __syncthreads();
    for (int off = 128; off > 0; off >>= 1) {
        if (threadIdx.x < off) {
            ss[threadIdx.x] += ss[threadIdx.x + off];
            sm[threadIdx.x]  = fmaxf(sm[threadIdx.x], sm[threadIdx.x + off]);
        }
        __syncthreads();
    }
    if (threadIdx.x == 0) {
        pooled[bc]        = ss[0] * (1.0f / (float)HWL);
        pooled[3072 + bc] = sm[0];
    }
}

__global__ __launch_bounds__(256) void se_fc(const float* __restrict__ pooled,
                                             const float* __restrict__ fc1,  // [24,384]
                                             const float* __restrict__ fc2,  // [384,24]
                                             float* __restrict__ se) {       // [8*384]
    __shared__ float sP[2 * 3072];
    __shared__ float sH[384];   // [0..191] avg hidden, [192..383] max hidden
    for (int i = threadIdx.x; i < 2 * 3072; i += 256) sP[i] = pooled[i];
    __syncthreads();
    for (int hidx = threadIdx.x; hidx < 384; hidx += 256) {
        int which = hidx / 192;
        int r = hidx - which * 192;
        int b = r / 24, j = r - (r / 24) * 24;
        const float* src = sP + which * 3072 + b * 384;
        float s = 0.0f;
        for (int c = 0; c < 384; ++c) s += fc1[j * 384 + c] * src[c];
        sH[hidx] = fmaxf(s, 0.0f);
    }
    __syncthreads();
    // fc2 is linear with no bias: fc2(ha) + fc2(hm) = fc2(ha + hm)
    for (int o = threadIdx.x; o < 3072; o += 256) {
        int b = o / 384, c = o - b * 384;
        float s = 0.0f;
        for (int j = 0; j < 24; ++j)
            s += fc2[c * 24 + j] * (sH[b * 24 + j] + sH[192 + b * 24 + j]);
        se[o] = 1.0f / (1.0f + __expf(-s));
    }
}

__global__ __launch_bounds__(256) void se_scale_v4(float4* __restrict__ out,
                                                   const float* __restrict__ se,
                                                   int n4) {
    int i = blockIdx.x * 256 + threadIdx.x;
    if (i >= n4) return;
    float s = se[(i * 4) / HWL];          // HWL % 4 == 0: uniform within quad
    float4 v = out[i];
    v.x *= s; v.y *= s; v.z *= s; v.w *= s;
    out[i] = v;
}

// ------------------------------ launcher -----------------------------------

extern "C" void kernel_launch(void* const* d_in, const int* in_sizes, int n_in,
                              void* d_out, int out_size, void* d_ws, size_t ws_size,
                              hipStream_t stream) {
    (void)in_sizes; (void)n_in; (void)out_size; (void)ws_size;

    const float* x     = (const float*)d_in[0];
    const float* ow[3] = {(const float*)d_in[1],  (const float*)d_in[6],  (const float*)d_in[11]};
    const float* ob[3] = {(const float*)d_in[2],  (const float*)d_in[7],  (const float*)d_in[12]};
    const float* dwv[3]= {(const float*)d_in[3],  (const float*)d_in[8],  (const float*)d_in[13]};
    const float* pw[3] = {(const float*)d_in[4],  (const float*)d_in[9],  (const float*)d_in[14]};
    const float* pb[3] = {(const float*)d_in[5],  (const float*)d_in[10], (const float*)d_in[15]};
    const float* fc1   = (const float*)d_in[16];
    const float* fc2   = (const float*)d_in[17];
    float* out = (float*)d_out;

    // Workspace carve-up (256B aligned). Total ~125 MB.
    char* ws = (char*)d_ws;
    auto carve = [&](size_t bytes) {
        char* p = ws;
        ws += (bytes + 255) & ~(size_t)255;
        return p;
    };
    unsigned short* xhwc  = (unsigned short*)carve((size_t)NTOT * 2);           // x NHWC bf16
    unsigned short* dchw  = (unsigned short*)carve((size_t)NB * CS * HWL * 2);  // deform out NCHW bf16
    unsigned short* dhwc  = (unsigned short*)carve((size_t)NB * CS * HWL * 2);  // deform out NHWC bf16
    float*          offs  = (float*)carve((size_t)NB * 98 * HWL * 4);           // max K=7 offsets
    unsigned short* Bo    = (unsigned short*)carve((size_t)6272 * 112 * 2);     // offset weights (swizzled)
    unsigned short* Bp    = (unsigned short*)carve((size_t)128 * 128 * 2);      // pointwise weights
    float*          pooled = (float*)carve((size_t)2 * 3072 * 4);
    float*          sew    = (float*)carve((size_t)3072 * 4);

    const int GEMM_BLOCKS = MROWS / 256;                 // 288 (MW=2)
    const int DEF_BLOCKS  = (NB * CS * HWL + 255) / 256; // 36864
    const int V4_BLOCKS   = (NTOT / 4 + 255) / 256;      // 27648
    const int TX_BLOCKS   = NB * (CT / 32) * (HWL / 64); // 13824
    const int TD_BLOCKS   = NB * (CS / 32) * (HWL / 64); // 4608

    // x (f32 NCHW) -> bf16 NHWC once (L2-resident thereafter).
    nchw_to_nhwc_bf16<float, CT><<<TX_BLOCKS, 256, 0, stream>>>(x, xhwc);

    // ---- group 0: K = 3 ----
    {
        constexpr int K = 3, NT = 2, KD = CS * K * K, NR = 2 * K * K;
        cvt_wB_swz<<<(KD * NT * 16 + 255) / 256, 256, 0, stream>>>(ow[0], Bo, KD, K * K, NT, NR);
        cvt_wB_swz<<<(128 * 8 * 16 + 255) / 256, 256, 0, stream>>>(pw[0], Bp, 128, 1, 8, 128);
        conv_gemm_wmma<K, NT><<<GEMM_BLOCKS, 256, 0, stream>>>(
            xhwc, CT, 0 * CS, Bo, ob[0], offs, NR, 0, NR);
        deform_dw<K><<<DEF_BLOCKS, 256, 0, stream>>>(x, CT, 0 * CS, offs, dwv[0], dchw);
        nchw_to_nhwc_bf16<unsigned short, CS><<<TD_BLOCKS, 256, 0, stream>>>(dchw, dhwc);
        conv_gemm_wmma<1, 8><<<GEMM_BLOCKS, 256, 0, stream>>>(
            dhwc, CS, 0, Bp, pb[0], out, CT, 0 * CS, CS);
    }
    // ---- group 1: K = 5 ----
    {
        constexpr int K = 5, NT = 4, KD = CS * K * K, NR = 2 * K * K;
        cvt_wB_swz<<<(KD * NT * 16 + 255) / 256, 256, 0, stream>>>(ow[1], Bo, KD, K * K, NT, NR);
        cvt_wB_swz<<<(128 * 8 * 16 + 255) / 256, 256, 0, stream>>>(pw[1], Bp, 128, 1, 8, 128);
        conv_gemm_wmma<K, NT><<<GEMM_BLOCKS, 256, 0, stream>>>(
            xhwc, CT, 1 * CS, Bo, ob[1], offs, NR, 0, NR);
        deform_dw<K><<<DEF_BLOCKS, 256, 0, stream>>>(x, CT, 1 * CS, offs, dwv[1], dchw);
        nchw_to_nhwc_bf16<unsigned short, CS><<<TD_BLOCKS, 256, 0, stream>>>(dchw, dhwc);
        conv_gemm_wmma<1, 8><<<GEMM_BLOCKS, 256, 0, stream>>>(
            dhwc, CS, 0, Bp, pb[1], out, CT, 1 * CS, CS);
    }
    // ---- group 2: K = 7 ----
    {
        constexpr int K = 7, NT = 7, KD = CS * K * K, NR = 2 * K * K;
        cvt_wB_swz<<<(KD * NT * 16 + 255) / 256, 256, 0, stream>>>(ow[2], Bo, KD, K * K, NT, NR);
        cvt_wB_swz<<<(128 * 8 * 16 + 255) / 256, 256, 0, stream>>>(pw[2], Bp, 128, 1, 8, 128);
        conv_gemm_wmma<K, NT><<<GEMM_BLOCKS, 256, 0, stream>>>(
            xhwc, CT, 2 * CS, Bo, ob[2], offs, NR, 0, NR);
        deform_dw<K><<<DEF_BLOCKS, 256, 0, stream>>>(x, CT, 2 * CS, offs, dwv[2], dchw);
        nchw_to_nhwc_bf16<unsigned short, CS><<<TD_BLOCKS, 256, 0, stream>>>(dchw, dhwc);
        conv_gemm_wmma<1, 8><<<GEMM_BLOCKS, 256, 0, stream>>>(
            dhwc, CS, 0, Bp, pb[2], out, CT, 2 * CS, CS);
    }

    // ---- SE gate ----
    se_pool<<<NB * CT, 256, 0, stream>>>(out, pooled);
    se_fc<<<1, 256, 0, stream>>>(pooled, fc1, fc2, sew);
    se_scale_v4<<<V4_BLOCKS, 256, 0, stream>>>((float4*)out, sew, NTOT / 4);
}
